// ARTabPFN_17171279249713
// MI455X (gfx1250) — matrix operand order repaired and checked by reading, not compiled, not run on hardware
//
#include <hip/hip_runtime.h>
#include <hip/hip_bf16.h>
#include <math.h>

typedef __bf16 bf16_t;
typedef __attribute__((ext_vector_type(16))) __bf16 v16bf;
typedef __attribute__((ext_vector_type(8)))  __bf16 v8bf;
typedef __attribute__((ext_vector_type(8)))  float   v8f;

#define NCTX 1024
#define NBUF 8
#define NTGT 512
#define RTOK 1544          // NCTX+NBUF+NTGT
#define DIM  512
#define NH   8
#define DHD  64
#define DFFD 1024
#define BB_  2
#define MTOK 3088          // BB_*RTOK
#define MA   3200          // padded rows (multiple of 128, >= attn read reach 3111)
#define RP   1568          // padded key count for Vt (multiple of 32, > 1567)
#define KK   5

// ---------------- WMMA fragment helpers (per CDNA5 ISA layouts) ----------------

// A-matrix 16x32 bf16: lane L holds row (L&15); elements 0..7 = K = kh*8..kh*8+7,
// elements 8..15 = K = 16+kh*8.., kh = L>>4.  Two 16B chunks at +0 and +16 elems.
__device__ inline v16bf load_frag_A(const bf16_t* base, int ld, int row0, int k0) {
  const int lane = threadIdx.x & 31;
  const bf16_t* p = base + (size_t)(row0 + (lane & 15)) * ld + k0 + ((lane >> 4) << 3);
  union { v16bf v; struct { v8bf lo, hi; } s; } u;
  u.s.lo = *(const v8bf*)p;
  u.s.hi = *(const v8bf*)(p + 16);
  return u.v;
}

// B-matrix 32x16 bf16 loaded from Bt = (N x K) row-major: lane L holds col (L&15);
// elements 0..7 = K = kh*8.., elements 8..15 = K = 16+kh*8.., kh = L>>4. Chunks at +0 and +16.
__device__ inline v16bf load_frag_B(const bf16_t* Bt, int ld, int n0, int k0) {
  const int lane = threadIdx.x & 31;
  const bf16_t* p = Bt + (size_t)(n0 + (lane & 15)) * ld + k0 + ((lane >> 4) << 3);
  union { v16bf v; struct { v8bf lo, hi; } s; } u;
  u.s.lo = *(const v8bf*)p;
  u.s.hi = *(const v8bf*)(p + 16);
  return u.v;
}

__device__ inline v8f wmma_bf16(v16bf a, v16bf b, v8f c) {
  return __builtin_amdgcn_wmma_f32_16x16x32_bf16(false, a, false, b, (short)0, c, false, false);
}

__device__ inline float gelu_exact(float x) {
  return 0.5f * x * (1.0f + erff(x * 0.70710678118654752f));
}

// analytic row mask (build_row_mask)
__device__ inline bool row_mask(int i, int j) {
  if (j >= RTOK) return false;
  if (j < NCTX) return true;
  if (i == j) return true;
  if (j >= NCTX + NBUF) return false;        // j is target: only diagonal
  if (i >= NCTX + NBUF) return true;         // target attends all buffer
  if (i >= NCTX) return j <= i;              // buffer causal over buffer
  return false;
}

// ---------------- kernels ----------------

// f32 (K x N) -> bf16 transposed (N x K)
__global__ void __launch_bounds__(256)
conv_t_kernel(const float* __restrict__ W, bf16_t* __restrict__ Wt, int Kd, int N) {
  int i = blockIdx.x * 256 + threadIdx.x;
  if (i < Kd * N) {
    int k = i / N, n = i - k * N;
    Wt[(size_t)n * Kd + k] = (bf16_t)W[i];
  }
}

// token embedding: one block per token row
__global__ void __launch_bounds__(256)
embed_kernel(const float* __restrict__ xc, const float* __restrict__ yc,
             const float* __restrict__ xbuf, const float* __restrict__ ybuf,
             const float* __restrict__ xt,
             const float* __restrict__ xW, const float* __restrict__ xB,
             const float* __restrict__ yW, const float* __restrict__ yB,
             const float* __restrict__ marker, const float* __restrict__ ar,
             float* __restrict__ X, bf16_t* __restrict__ Xb) {
  int row = blockIdx.x;                 // 0..MTOK-1
  int b = row / RTOK, r = row - b * RTOK;
  const float* xrow; const float* mk; const float* arrow = nullptr;
  float yval = 0.f; int hasY = 0;
  if (r < NCTX)            { xrow = xc   + (size_t)(b * NCTX + r) * 32;            yval = yc[b * NCTX + r];   mk = marker + DIM;     hasY = 1; }
  else if (r < NCTX + NBUF){ int q = r - NCTX; xrow = xbuf + (size_t)(b * NBUF + q) * 32; yval = ybuf[b * NBUF + q]; mk = marker + 2 * DIM; hasY = 1; arrow = ar + (size_t)q * DIM; }
  else                     { int q = r - NCTX - NBUF; xrow = xt + (size_t)(b * NTGT + q) * 32; mk = marker; }
  __shared__ float sm;
  float v = (threadIdx.x < 32) ? xrow[threadIdx.x] : 0.f;
  #pragma unroll
  for (int o = 16; o > 0; o >>= 1) v += __shfl_down(v, o, 32);
  if (threadIdx.x == 0) sm = v * (1.0f / 32.0f);
  __syncthreads();
  float xm = sm;
  for (int d = threadIdx.x; d < DIM; d += 256) {
    float e = xm * xW[d] + xB[d] + mk[d];
    if (hasY)  e += yval * yW[d] + yB[d];
    if (arrow) e += arrow[d];
    X[(size_t)row * DIM + d] = e;
    Xb[(size_t)row * DIM + d] = (bf16_t)e;
  }
}

// generic WMMA GEMM: C = A(MxK) * Bt(NxK)^T + bias, optional GELU,
// outputs f32 C and/or bf16 Cb, or per-head-transposed Vt (outmode=1).
// 4 waves / 128 threads per block; each wave owns a 64x64 register tile
// (16 accumulators, 4x operand reuse -> 16 WMMA per 8 fragment loads).
__global__ void __launch_bounds__(128)
gemm_wmma(const bf16_t* __restrict__ A, const bf16_t* __restrict__ Bt,
          const float* __restrict__ bias, int nbias,
          float* __restrict__ C, bf16_t* __restrict__ Cb, bf16_t* __restrict__ Vt,
          int M, int N, int Kd, int lda, int ldb, int ldc, int act, int outmode) {
  const int w = threadIdx.x >> 5;          // 0..3
  const int lane = threadIdx.x & 31;
  const int m0 = blockIdx.y * 128 + (w & 1) * 64;
  const int n0 = blockIdx.x * 128 + (w >> 1) * 64;
  v8f zero = {0.f, 0.f, 0.f, 0.f, 0.f, 0.f, 0.f, 0.f};
  v8f acc[4][4];
  #pragma unroll
  for (int i = 0; i < 4; i++)
    #pragma unroll
    for (int j = 0; j < 4; j++) acc[i][j] = zero;

  for (int k = 0; k < Kd; k += 32) {
    // prefetch two K-steps ahead (stays inside workspace; speculative-safe)
    __builtin_prefetch((const void*)(A  + (size_t)(m0 + (lane & 15)) * lda + k + 64), 0, 3);
    __builtin_prefetch((const void*)(Bt + (size_t)(n0 + (lane & 15)) * ldb + k + 64), 0, 3);
    v16bf a[4], b[4];
    #pragma unroll
    for (int i = 0; i < 4; i++) a[i] = load_frag_A(A, lda, m0 + 16 * i, k);
    #pragma unroll
    for (int j = 0; j < 4; j++) b[j] = load_frag_B(Bt, ldb, n0 + 16 * j, k);
    #pragma unroll
    for (int i = 0; i < 4; i++)
      #pragma unroll
      for (int j = 0; j < 4; j++)
        acc[i][j] = wmma_bf16(a[i], b[j], acc[i][j]);
  }

  const int col = lane & 15, hi = lane >> 4;
  #pragma unroll
  for (int i = 0; i < 4; i++) {
    #pragma unroll
    for (int j = 0; j < 4; j++) {
      #pragma unroll
      for (int v = 0; v < 8; v++) {
        int m = m0 + i * 16 + v + 8 * hi;
        int n = n0 + j * 16 + col;
        float val = acc[i][j][v] + ((n < nbias) ? bias[n] : 0.f);
        if (act) val = gelu_exact(val);
        if (outmode == 0) {
          if (m < M && n < N) {
            if (C)  C[(size_t)m * ldc + n] = val;
            if (Cb) Cb[(size_t)m * ldc + n] = (bf16_t)val;
          }
        } else {
          if (m < M) {
            int bb = m / RTOK, r = m - bb * RTOK;
            int hh = n >> 6, cc = n & 63;
            Vt[((size_t)((bb * NH + hh) * DHD + cc)) * RP + r] = (bf16_t)val;
          }
        }
      }
    }
  }
}

// fused residual-add + LayerNorm (row per block); writes f32 X and bf16 Xb
__global__ void __launch_bounds__(256)
add_ln_kernel(const float* __restrict__ Y, const float* __restrict__ Rres,
              const float* __restrict__ g, const float* __restrict__ be,
              float* __restrict__ X, bf16_t* __restrict__ Xb) {
  int row = blockIdx.x;
  int t = threadIdx.x;
  size_t base = (size_t)row * DIM;
  float a = Y[base + t] + Rres[base + t];
  float b = Y[base + t + 256] + Rres[base + t + 256];
  __shared__ float red[256];
  red[t] = a + b; __syncthreads();
  for (int o = 128; o > 0; o >>= 1) { if (t < o) red[t] += red[t + o]; __syncthreads(); }
  float mu = red[0] * (1.0f / DIM);
  __syncthreads();
  float da = a - mu, db = b - mu;
  red[t] = da * da + db * db; __syncthreads();
  for (int o = 128; o > 0; o >>= 1) { if (t < o) red[t] += red[t + o]; __syncthreads(); }
  float rs = rsqrtf(red[0] * (1.0f / DIM) + 1e-5f);
  float o0 = da * rs * g[t] + be[t];
  float o1 = db * rs * g[t + 256] + be[t + 256];
  X[base + t] = o0;       Xb[base + t] = (bf16_t)o0;
  X[base + t + 256] = o1; Xb[base + t + 256] = (bf16_t)o1;
}

// final LayerNorm on target rows only, gathered to compact bf16 (1024 x 512)
__global__ void __launch_bounds__(256)
final_ln_gather(const float* __restrict__ X, const float* __restrict__ g,
                const float* __restrict__ be, bf16_t* __restrict__ Hb) {
  int i = blockIdx.x;                     // 0..1023
  int b = i >> 9, r = NCTX + NBUF + (i & 511);
  size_t src = ((size_t)b * RTOK + r) * DIM;
  int t = threadIdx.x;
  float a = X[src + t], bb = X[src + t + 256];
  __shared__ float red[256];
  red[t] = a + bb; __syncthreads();
  for (int o = 128; o > 0; o >>= 1) { if (t < o) red[t] += red[t + o]; __syncthreads(); }
  float mu = red[0] * (1.0f / DIM);
  __syncthreads();
  float da = a - mu, db = bb - mu;
  red[t] = da * da + db * db; __syncthreads();
  for (int o = 128; o > 0; o >>= 1) { if (t < o) red[t] += red[t + o]; __syncthreads(); }
  float rs = rsqrtf(red[0] * (1.0f / DIM) + 1e-5f);
  Hb[(size_t)i * DIM + t]       = (bf16_t)(da * rs * g[t] + be[t]);
  Hb[(size_t)i * DIM + t + 256] = (bf16_t)(db * rs * g[t + 256] + be[t + 256]);
}

// flash-attention over rows: one wave = 16 queries of one (batch, head)
#define PSTRIDE 56
__global__ void __launch_bounds__(256)
attn_kernel(const bf16_t* __restrict__ Qb, const bf16_t* __restrict__ Kb,
            const bf16_t* __restrict__ Vt, bf16_t* __restrict__ AOb) {
  const int w = threadIdx.x >> 5;
  const int lane = threadIdx.x & 31;
  const int col = lane & 15, hi = lane >> 4;
  const int b = blockIdx.z, h = blockIdx.y;
  const int q0 = (blockIdx.x * 8 + w) * 16;
  if (q0 >= 1552) return;                 // wave-uniform exit, no barriers used
  __shared__ bf16_t pbuf[8][16 * PSTRIDE];
  bf16_t* pw = &pbuf[w][0];
  const bf16_t* Qbase = Qb + (size_t)b * RTOK * DIM + h * DHD;
  const bf16_t* Kbase = Kb + (size_t)b * RTOK * DIM + h * DHD;
  const bf16_t* Vbase = Vt + (size_t)((b * NH + h) * DHD) * RP;
  v16bf qf0 = load_frag_A(Qbase, DIM, q0, 0);
  v16bf qf1 = load_frag_A(Qbase, DIM, q0, 32);
  v8f zero = {0.f, 0.f, 0.f, 0.f, 0.f, 0.f, 0.f, 0.f};
  v8f o0 = zero, o1 = zero, o2 = zero, o3 = zero;
  float mrow[8], lrow[8];
  #pragma unroll
  for (int v = 0; v < 8; v++) { mrow[v] = -1e30f; lrow[v] = 0.f; }

  for (int k0 = 0; k0 < RTOK; k0 += 32) {
    v8f s0 = zero, s1 = zero;
    v16bf kf;
    kf = load_frag_B(Kbase, DIM, k0,      0);  s0 = wmma_bf16(qf0, kf, s0);
    kf = load_frag_B(Kbase, DIM, k0,      32); s0 = wmma_bf16(qf1, kf, s0);
    kf = load_frag_B(Kbase, DIM, k0 + 16, 0);  s1 = wmma_bf16(qf0, kf, s1);
    kf = load_frag_B(Kbase, DIM, k0 + 16, 32); s1 = wmma_bf16(qf1, kf, s1);
    #pragma unroll
    for (int v = 0; v < 8; v++) {
      int qr = q0 + v + 8 * hi;
      float a = row_mask(qr, k0 + col)      ? s0[v] * 0.125f : -1e9f;
      float c = row_mask(qr, k0 + 16 + col) ? s1[v] * 0.125f : -1e9f;
      s0[v] = a; s1[v] = c;
      float t = fmaxf(a, c);
      t = fmaxf(t, __shfl_xor(t, 1));
      t = fmaxf(t, __shfl_xor(t, 2));
      t = fmaxf(t, __shfl_xor(t, 4));
      t = fmaxf(t, __shfl_xor(t, 8));       // row max within half-wave
      float mn = fmaxf(mrow[v], t);
      float alpha = __expf(mrow[v] - mn);
      mrow[v] = mn;
      float p0 = __expf(s0[v] - mn);
      float p1 = __expf(s1[v] - mn);
      float rs = p0 + p1;
      rs += __shfl_xor(rs, 1);
      rs += __shfl_xor(rs, 2);
      rs += __shfl_xor(rs, 4);
      rs += __shfl_xor(rs, 8);
      lrow[v] = lrow[v] * alpha + rs;
      o0[v] *= alpha; o1[v] *= alpha; o2[v] *= alpha; o3[v] *= alpha;
      int prow = (v + 8 * hi) * PSTRIDE;
      pw[prow + col]      = (bf16_t)p0;
      pw[prow + 16 + col] = (bf16_t)p1;
    }
    v16bf pf = load_frag_A(pw, PSTRIDE, 0, 0);  // P tile 16x32, A layout
    v16bf vf;
    vf = load_frag_B(Vbase, RP, 0,  k0); o0 = wmma_bf16(pf, vf, o0);
    vf = load_frag_B(Vbase, RP, 16, k0); o1 = wmma_bf16(pf, vf, o1);
    vf = load_frag_B(Vbase, RP, 32, k0); o2 = wmma_bf16(pf, vf, o2);
    vf = load_frag_B(Vbase, RP, 48, k0); o3 = wmma_bf16(pf, vf, o3);
  }
  #pragma unroll
  for (int v = 0; v < 8; v++) {
    int qr = q0 + v + 8 * hi;
    if (qr < RTOK) {
      float inv = 1.0f / lrow[v];
      size_t rowoff = ((size_t)(b * RTOK + qr)) * DIM + h * DHD;
      AOb[rowoff + col]      = (bf16_t)(o0[v] * inv);
      AOb[rowoff + 16 + col] = (bf16_t)(o1[v] * inv);
      AOb[rowoff + 32 + col] = (bf16_t)(o2[v] * inv);
      AOb[rowoff + 48 + col] = (bf16_t)(o3[v] * inv);
    }
  }
}

// mixture-of-gaussians head: per (b,t) row; writes mean/std/w to d_out, ll to scratch
__global__ void __launch_bounds__(256)
mixture_kernel(const float* __restrict__ H2, const float* __restrict__ yt,
               const float* __restrict__ mb, const float* __restrict__ sb,
               const float* __restrict__ wb, float* __restrict__ out,
               float* __restrict__ llbuf) {
  int i = blockIdx.x * 256 + threadIdx.x;
  if (i >= BB_ * NTGT) return;
  float y = yt[i];
  float means[KK], stds[KK], wl[KK];
  float wmax = -1e30f;
  #pragma unroll
  for (int k = 0; k < KK; k++) {
    float m  = H2[i * 16 + 3 * k]     + mb[k];
    float sr = H2[i * 16 + 3 * k + 1] + sb[k];
    float sp = (sr > 20.f) ? sr : log1pf(__expf(sr));
    float s  = fminf(sp, 2.0f) + 0.001f;
    float wv = H2[i * 16 + 3 * k + 2] + wb[k];
    means[k] = m; stds[k] = s; wl[k] = wv;
    wmax = fmaxf(wmax, wv);
  }
  float wsum = 0.f;
  #pragma unroll
  for (int k = 0; k < KK; k++) wsum += __expf(wl[k] - wmax);
  float lse_w = wmax + logf(wsum);
  const float LOG2PI = 1.8378770664093453f;
  float lp[KK], lmax = -1e30f;
  #pragma unroll
  for (int k = 0; k < KK; k++) {
    float logw = fmaxf(wl[k] - lse_w, -27.63102112f);   // log(1e-12)
    float z = (y - means[k]) / stds[k];
    lp[k] = -0.5f * (LOG2PI + 2.f * logf(stds[k]) + z * z) + logw;
    lmax = fmaxf(lmax, lp[k]);
  }
  float es = 0.f;
  #pragma unroll
  for (int k = 0; k < KK; k++) es += __expf(lp[k] - lmax);
  llbuf[i] = lmax + logf(es);
  #pragma unroll
  for (int k = 0; k < KK; k++) {
    out[1 + i * KK + k]                       = means[k];
    out[1 + BB_ * NTGT * KK + i * KK + k]     = stds[k];
    out[1 + 2 * BB_ * NTGT * KK + i * KK + k] = __expf(wl[k] - lse_w);
  }
}

__global__ void __launch_bounds__(256)
loss_final_kernel(const float* __restrict__ llbuf, float* __restrict__ out) {
  __shared__ float red[256];
  float s = 0.f;
  for (int i = threadIdx.x; i < BB_ * NTGT; i += 256) s += llbuf[i];
  red[threadIdx.x] = s; __syncthreads();
  for (int o = 128; o > 0; o >>= 1) { if (threadIdx.x < o) red[threadIdx.x] += red[threadIdx.x + o]; __syncthreads(); }
  if (threadIdx.x == 0) out[0] = -red[0] / (float)(BB_ * NTGT);
}

// ---------------- host orchestration ----------------

extern "C" void kernel_launch(void* const* d_in, const int* in_sizes, int n_in,
                              void* d_out, int out_size, void* d_ws, size_t ws_size,
                              hipStream_t stream) {
  (void)in_sizes; (void)n_in; (void)out_size; (void)ws_size;
  // leaf index map (setup_inputs insertion order)
  auto F = [&](int i) -> const float* { return (const float*)d_in[i]; };
  const int LBASE = 6, LSTRIDE = 26;

  // workspace carve
  size_t off = 0;
  auto carve = [&](size_t bytes) -> char* {
    char* p = (char*)d_ws + off;
    off += (bytes + 255) & ~(size_t)255;
    return p;
  };
  float*  X    = (float*) carve((size_t)MA * DIM * 4);
  bf16_t* Xb   = (bf16_t*)carve((size_t)MA * DIM * 2);
  float*  T1   = (float*) carve((size_t)MA * DIM * 4);
  bf16_t* T1b  = (bf16_t*)carve((size_t)MA * DIM * 2);
  float*  T0   = (float*) carve((size_t)MA * DFFD * 4);
  bf16_t* T0b  = (bf16_t*)carve((size_t)MA * DFFD * 2);
  bf16_t* Qb   = (bf16_t*)carve((size_t)MA * DIM * 2);
  bf16_t* Kbf  = (bf16_t*)carve((size_t)MA * DIM * 2);
  bf16_t* Vt   = (bf16_t*)carve((size_t)BB_ * NH * DHD * RP * 2);
  bf16_t* wt_fv  = (bf16_t*)carve((size_t)DIM * DIM * 2);
  bf16_t* wt_fo  = (bf16_t*)carve((size_t)DIM * DIM * 2);
  bf16_t* wt_rq  = (bf16_t*)carve((size_t)DIM * DIM * 2);
  bf16_t* wt_rk  = (bf16_t*)carve((size_t)DIM * DIM * 2);
  bf16_t* wt_rv  = (bf16_t*)carve((size_t)DIM * DIM * 2);
  bf16_t* wt_ro  = (bf16_t*)carve((size_t)DIM * DIM * 2);
  bf16_t* wt_ff1 = (bf16_t*)carve((size_t)DFFD * DIM * 2);
  bf16_t* wt_ff2 = (bf16_t*)carve((size_t)DIM * DFFD * 2);
  bf16_t* wt_h1  = (bf16_t*)carve((size_t)DFFD * DIM * 2);
  bf16_t* wt_h2  = (bf16_t*)carve((size_t)128 * DFFD * 2);   // 15 rows used, rest zero
  bf16_t* Hinb   = (bf16_t*)carve((size_t)BB_ * NTGT * DIM * 2);
  float*  H2     = (float*) carve((size_t)BB_ * NTGT * 16 * 4);
  float*  llbuf  = (float*) carve((size_t)BB_ * NTGT * 4);

  hipMemsetAsync(d_ws, 0, off, stream);

  auto conv_t = [&](const float* W, bf16_t* Wt, int Kd, int N) {
    int tot = Kd * N;
    conv_t_kernel<<<(tot + 255) / 256, 256, 0, stream>>>(W, Wt, Kd, N);
  };
  // Npad must be a multiple of 128 (block tile); Mcover a multiple of 128
  auto gemm = [&](const bf16_t* A, const bf16_t* Bt, const float* bias, int nbias,
                  float* C, bf16_t* Cb, bf16_t* Vto, int M, int Mcover, int Npad,
                  int N, int Kd, int lda, int ldb, int ldc, int act, int outmode) {
    dim3 g(Npad / 128, Mcover / 128);
    gemm_wmma<<<g, 128, 0, stream>>>(A, Bt, bias, nbias, C, Cb, Vto,
                                     M, N, Kd, lda, ldb, ldc, act, outmode);
  };

  // embedding
  embed_kernel<<<MTOK, 256, 0, stream>>>(
      F(119), F(120), F(121), F(122), F(123),
      F(0), F(1), F(2), F(3), F(4), F(5), X, Xb);

  for (int l = 0; l < 4; l++) {
    int lb = LBASE + l * LSTRIDE;
    // weight transposes (f32 -> bf16^T), L2-resident afterwards
    conv_t(F(lb + 4),  wt_fv, DIM, DIM);
    conv_t(F(lb + 6),  wt_fo, DIM, DIM);
    conv_t(F(lb + 8),  wt_rq, DIM, DIM);
    conv_t(F(lb + 10), wt_rk, DIM, DIM);
    conv_t(F(lb + 12), wt_rv, DIM, DIM);
    conv_t(F(lb + 14), wt_ro, DIM, DIM);
    conv_t(F(lb + 16), wt_ff1, DIM, DFFD);
    conv_t(F(lb + 18), wt_ff2, DFFD, DIM);

    // attn_f with C=1 degenerates to (x Wv + bv) Wo + bo
    gemm(Xb,  wt_fv, F(lb + 5), DIM, nullptr, T1b, nullptr, MTOK, MA, DIM, DIM, DIM, DIM, DIM, DIM, 0, 0);
    gemm(T1b, wt_fo, F(lb + 7), DIM, T0, nullptr, nullptr,  MTOK, MA, DIM, DIM, DIM, DIM, DIM, DIM, 0, 0);
    add_ln_kernel<<<MTOK, 256, 0, stream>>>(T0, X, F(lb + 20), F(lb + 21), X, Xb);

    // row attention
    gemm(Xb, wt_rq, F(lb + 9),  DIM, nullptr, Qb,  nullptr, MTOK, MA, DIM, DIM, DIM, DIM, DIM, DIM, 0, 0);
    gemm(Xb, wt_rk, F(lb + 11), DIM, nullptr, Kbf, nullptr, MTOK, MA, DIM, DIM, DIM, DIM, DIM, DIM, 0, 0);
    gemm(Xb, wt_rv, F(lb + 13), DIM, nullptr, nullptr, Vt,  MTOK, MA, DIM, DIM, DIM, DIM, DIM, DIM, 0, 1);
    attn_kernel<<<dim3(13, NH, BB_), 256, 0, stream>>>(Qb, Kbf, Vt, T1b);
    gemm(T1b, wt_ro, F(lb + 15), DIM, T0, nullptr, nullptr, MTOK, MA, DIM, DIM, DIM, DIM, DIM, DIM, 0, 0);
    add_ln_kernel<<<MTOK, 256, 0, stream>>>(T0, X, F(lb + 22), F(lb + 23), X, Xb);

    // feed-forward
    gemm(Xb,  wt_ff1, F(lb + 17), DFFD, T0, T0b, nullptr, MTOK, MA, DFFD, DFFD, DIM,  DIM,  DIM,  DFFD, 1, 0);
    gemm(T0b, wt_ff2, F(lb + 19), DIM,  T1, nullptr, nullptr, MTOK, MA, DIM, DIM, DFFD, DFFD, DFFD, DIM, 0, 0);
    add_ln_kernel<<<MTOK, 256, 0, stream>>>(T1, X, F(lb + 24), F(lb + 25), X, Xb);
  }

  // head
  conv_t(F(112), wt_h1, DIM, DFFD);
  conv_t(F(114), wt_h2, DFFD, 15);
  final_ln_gather<<<BB_ * NTGT, 256, 0, stream>>>(X, F(110), F(111), Hinb);
  gemm(Hinb, wt_h1, F(113), DFFD, T0, T0b, nullptr, BB_ * NTGT, BB_ * NTGT, DFFD, DFFD, DIM, DIM, DIM, DFFD, 1, 0);
  gemm(T0b,  wt_h2, F(115), 15,   H2, nullptr, nullptr, BB_ * NTGT, BB_ * NTGT, 128, 16, DFFD, DFFD, DFFD, 16, 0, 0);

  mixture_kernel<<<(BB_ * NTGT + 255) / 256, 256, 0, stream>>>(
      H2, F(124), F(116), F(117), F(118), (float*)d_out, llbuf);
  loss_final_kernel<<<1, 256, 0, stream>>>(llbuf, (float*)d_out);
}
